// DRQN_20323785245315
// MI455X (gfx1250) — matrix operand order repaired and checked by reading, not compiled, not run on hardware
//
#include <hip/hip_runtime.h>
#include <hip/hip_bf16.h>

typedef __bf16 v16bf __attribute__((ext_vector_type(16)));
typedef __bf16 v8bf  __attribute__((ext_vector_type(8)));
typedef float  v8f   __attribute__((ext_vector_type(8)));
typedef float  f4    __attribute__((ext_vector_type(4)));

#define OBS 512
#define ACT 32
#define KIN 544   /* OBS+ACT */
#define EMB 512
#define HID 512
#define G4  2048  /* 4*HID */
#define NQ  32
#define TM  32    /* batch rows per block */

#define AST (KIN + 8)   /* Abuf row stride (bf16 elems), 16B-aligned rows */
#define XST (EMB + 8)   /* Xbuf/Hbuf row stride */

/* workspace layout (bytes) */
#define WS_WIH   0u
#define WS_W1    (G4 * EMB * 2u)                 /* after W_ih bf16 */
#define WS_W2    (WS_W1 + EMB * KIN * 2u)        /* after W1 bf16  */
#define WS_HVEC  (WS_W2 + NQ * HID * 2u)         /* after W2 bf16  */

__device__ __forceinline__ v8f wmma_bf16(v16bf a, v16bf b, v8f c) {
    return __builtin_amdgcn_wmma_f32_16x16x32_bf16(false, a, false, b, (short)0, c,
                                                   false, false);
}

/* A fragment (16x32 bf16, M x K): lane m=lane&15, hf=lane>>4.
   element e -> A[m][k0 + e + 8*hf + (e>=8 ? 8 : 0)] : two 16B chunks. */
__device__ __forceinline__ v16bf load_a_frag(const __bf16* rowbase, int k0, int hf) {
    v8bf lo = *(const v8bf*)(rowbase + k0 + 8 * hf);
    v8bf hi = *(const v8bf*)(rowbase + k0 + 16 + 8 * hf);
    return __builtin_shufflevector(lo, hi, 0, 1, 2, 3, 4, 5, 6, 7,
                                           8, 9, 10, 11, 12, 13, 14, 15);
}

/* B fragment (32x16 bf16, K x N) from W row-major [N][K] (B = W^T):
   lane n=lane&15, hf=lane>>4. element e -> W[n][k0 + e + 8*hf + (e>=8 ? 8 : 0)] : two 16B chunks. */
__device__ __forceinline__ v16bf load_b_frag(const __bf16* wrow, int k0, int hf) {
    v8bf lo = *(const v8bf*)(wrow + k0 + 8 * hf);
    v8bf hi = *(const v8bf*)(wrow + k0 + 16 + 8 * hf);
    return __builtin_shufflevector(lo, hi, 0, 1, 2, 3, 4, 5, 6, 7,
                                           8, 9, 10, 11, 12, 13, 14, 15);
}

__device__ __forceinline__ float sigmoidf_(float x) { return 1.0f / (1.0f + expf(-x)); }

/* ---------------- prep kernels ---------------- */

__global__ void cvt_bf16_kernel(const float* __restrict__ src, __bf16* __restrict__ dst, int n) {
    int i = blockIdx.x * blockDim.x + threadIdx.x;
    int stride = gridDim.x * blockDim.x;
    for (; i < n; i += stride) dst[i] = (__bf16)src[i];
}

/* hvec[n] = b_ih[n] + b_hh[n] + dot(W_hh[n,:], hidden)  (row-invariant gate term) */
__global__ void hvec_kernel(const float* __restrict__ Whh, const float* __restrict__ hidden,
                            const float* __restrict__ bih, const float* __restrict__ bhh,
                            float* __restrict__ hvec) {
    int n = blockIdx.x * blockDim.x + threadIdx.x;
    if (n >= G4) return;
    const f4* w = (const f4*)(Whh + (size_t)n * HID);
    const f4* h = (const f4*)hidden;
    float s = 0.0f;
#pragma unroll 4
    for (int k = 0; k < HID / 4; ++k) {
        f4 a = w[k], b = h[k];
        s += a.x * b.x + a.y * b.y + a.z * b.z + a.w * b.w;
    }
    hvec[n] = s + bih[n] + bhh[n];
}

/* ---------------- fused main kernel ---------------- */

__global__ __launch_bounds__(128)
void drqn_main(const float* __restrict__ obs, const float* __restrict__ pact,
               const float* __restrict__ cell, const float* __restrict__ b1,
               const float* __restrict__ b2,
               const __bf16* __restrict__ W1b, const __bf16* __restrict__ Wihb,
               const __bf16* __restrict__ W2b, const float* __restrict__ hvec,
               float* __restrict__ q_out, float* __restrict__ h_last,
               float* __restrict__ c_last, int nblocks)
{
    __shared__ __attribute__((aligned(16))) __bf16 Abuf[TM * AST];
    __shared__ __attribute__((aligned(16))) __bf16 Xbuf[TM * XST];
    __shared__ __attribute__((aligned(16))) __bf16 Hbuf[TM * XST];

    const int tid  = threadIdx.x;
    const int wave = tid >> 5;
    const int lane = tid & 31;
    const int hf   = lane >> 4;
    const int ln   = lane & 15;
    const long row0 = (long)blockIdx.x * TM;
    const bool lastblk = (blockIdx.x == (unsigned)(nblocks - 1));

    /* ---- Stage A: load 32 rows of concat(obs, act), convert to bf16 in LDS ---- */
    for (int idx = tid; idx < TM * (KIN / 4); idx += 128) {
        int r   = idx / (KIN / 4);
        int col = (idx % (KIN / 4)) * 4;
        f4 v;
        if (col < OBS) v = *(const f4*)(obs + (row0 + r) * OBS + col);
        else           v = *(const f4*)(pact + (row0 + r) * ACT + (col - OBS));
        __bf16* d = Abuf + r * AST + col;
        d[0] = (__bf16)v.x; d[1] = (__bf16)v.y; d[2] = (__bf16)v.z; d[3] = (__bf16)v.w;
    }
    __syncthreads();

    /* ---- GEMM1: X = relu(A @ W1^T + b1), bf16 into Xbuf ---- */
    {
        const __bf16* arow0 = Abuf + (size_t)ln * AST;
        const __bf16* arow1 = Abuf + (size_t)(ln + 16) * AST;
        for (int t = wave; t < EMB / 16; t += 4) {
            const int n = t * 16 + ln;
            const __bf16* wrow = W1b + (size_t)n * KIN;
            v8f acc0 = {}, acc1 = {};
            for (int k0 = 0; k0 < KIN; k0 += 32) {
                v16bf b  = load_b_frag(wrow, k0, hf);
                v16bf a0 = load_a_frag(arow0, k0, hf);
                v16bf a1 = load_a_frag(arow1, k0, hf);
                acc0 = wmma_bf16(a0, b, acc0);
                acc1 = wmma_bf16(a1, b, acc1);
            }
            float bias = b1[n];
#pragma unroll
            for (int r = 0; r < 8; ++r) {
                float x0 = acc0[r] + bias; x0 = x0 > 0.0f ? x0 : 0.0f;
                float x1 = acc1[r] + bias; x1 = x1 > 0.0f ? x1 : 0.0f;
                Xbuf[(r + 8 * hf) * XST + n]        = (__bf16)x0;
                Xbuf[(r + 8 * hf + 16) * XST + n]   = (__bf16)x1;
            }
        }
    }
    __syncthreads();

    /* ---- GEMM2 + LSTM elementwise: gates -> h into Hbuf (and last-row h/c) ---- */
    {
        const __bf16* arow0 = Xbuf + (size_t)ln * XST;
        const __bf16* arow1 = Xbuf + (size_t)(ln + 16) * XST;
        for (int t = wave; t < HID / 16; t += 4) {
            const int j = t * 16 + ln;
            const __bf16* pi = Wihb + (size_t)(j)           * EMB;
            const __bf16* pf = Wihb + (size_t)(j + HID)     * EMB;
            const __bf16* pg = Wihb + (size_t)(j + 2 * HID) * EMB;
            const __bf16* po = Wihb + (size_t)(j + 3 * HID) * EMB;
            v8f ai0 = {}, ai1 = {}, af0 = {}, af1 = {};
            v8f ag0 = {}, ag1 = {}, ao0 = {}, ao1 = {};
            for (int k0 = 0; k0 < EMB; k0 += 32) {
                v16bf a0 = load_a_frag(arow0, k0, hf);
                v16bf a1 = load_a_frag(arow1, k0, hf);
                v16bf bi = load_b_frag(pi, k0, hf);
                v16bf bf = load_b_frag(pf, k0, hf);
                v16bf bg = load_b_frag(pg, k0, hf);
                v16bf bo = load_b_frag(po, k0, hf);
                ai0 = wmma_bf16(a0, bi, ai0); ai1 = wmma_bf16(a1, bi, ai1);
                af0 = wmma_bf16(a0, bf, af0); af1 = wmma_bf16(a1, bf, af1);
                ag0 = wmma_bf16(a0, bg, ag0); ag1 = wmma_bf16(a1, bg, ag1);
                ao0 = wmma_bf16(a0, bo, ao0); ao1 = wmma_bf16(a1, bo, ao1);
            }
            const float hvi = hvec[j];
            const float hvf = hvec[j + HID];
            const float hvg = hvec[j + 2 * HID];
            const float hvo = hvec[j + 3 * HID];
            const float cp  = cell[j];
#pragma unroll
            for (int r = 0; r < 8; ++r) {
                /* M-tile 0: rows 0..15 */
                {
                    float iv = sigmoidf_(ai0[r] + hvi);
                    float fv = sigmoidf_(af0[r] + hvf);
                    float gv = tanhf(ag0[r] + hvg);
                    float ov = sigmoidf_(ao0[r] + hvo);
                    float c  = fv * cp + iv * gv;
                    float h  = ov * tanhf(c);
                    Hbuf[(r + 8 * hf) * XST + j] = (__bf16)h;
                }
                /* M-tile 1: rows 16..31 (row 31 = global row B-1 in last block) */
                {
                    float iv = sigmoidf_(ai1[r] + hvi);
                    float fv = sigmoidf_(af1[r] + hvf);
                    float gv = tanhf(ag1[r] + hvg);
                    float ov = sigmoidf_(ao1[r] + hvo);
                    float c  = fv * cp + iv * gv;
                    float h  = ov * tanhf(c);
                    Hbuf[(r + 8 * hf + 16) * XST + j] = (__bf16)h;
                    if (lastblk && r == 7 && hf == 1) {   /* m == 31 */
                        h_last[j] = h;
                        c_last[j] = c;
                    }
                }
            }
        }
    }
    __syncthreads();

    /* ---- GEMM3: Q = H @ W2^T + b2 ; 4 tiles, one per wave ---- */
    {
        const int nt = wave & 1;
        const int mt = wave >> 1;
        const int n  = nt * 16 + ln;
        const __bf16* wrow = W2b + (size_t)n * HID;
        const __bf16* arow = Hbuf + (size_t)(ln + 16 * mt) * XST;
        v8f acc = {};
        for (int k0 = 0; k0 < HID; k0 += 32) {
            v16bf a = load_a_frag(arow, k0, hf);
            v16bf b = load_b_frag(wrow, k0, hf);
            acc = wmma_bf16(a, b, acc);
        }
        const float bias = b2[n];
#pragma unroll
        for (int r = 0; r < 8; ++r) {
            long m = row0 + mt * 16 + r + 8 * hf;
            q_out[m * NQ + n] = acc[r] + bias;
        }
    }
}

/* ---------------- launcher ---------------- */

extern "C" void kernel_launch(void* const* d_in, const int* in_sizes, int n_in,
                              void* d_out, int out_size, void* d_ws, size_t ws_size,
                              hipStream_t stream) {
    const float* obs  = (const float*)d_in[0];
    const float* pact = (const float*)d_in[1];
    const float* hid  = (const float*)d_in[2];
    const float* cell = (const float*)d_in[3];
    const float* W1   = (const float*)d_in[4];
    const float* b1   = (const float*)d_in[5];
    const float* Wih  = (const float*)d_in[6];
    const float* bih  = (const float*)d_in[7];
    const float* Whh  = (const float*)d_in[8];
    const float* bhh  = (const float*)d_in[9];
    const float* W2   = (const float*)d_in[10];
    const float* b2   = (const float*)d_in[11];

    const int B = in_sizes[0] / OBS;

    char* ws = (char*)d_ws;
    __bf16* Wihb = (__bf16*)(ws + WS_WIH);
    __bf16* W1b  = (__bf16*)(ws + WS_W1);
    __bf16* W2b  = (__bf16*)(ws + WS_W2);
    float*  hvec = (float*)(ws + WS_HVEC);

    cvt_bf16_kernel<<<256, 256, 0, stream>>>(Wih, Wihb, G4 * EMB);
    cvt_bf16_kernel<<<128, 256, 0, stream>>>(W1, W1b, EMB * KIN);
    cvt_bf16_kernel<<<16, 256, 0, stream>>>(W2, W2b, NQ * HID);
    hvec_kernel<<<G4 / 256, 256, 0, stream>>>(Whh, hid, bih, bhh, hvec);

    float* q_out  = (float*)d_out;
    float* h_last = q_out + (size_t)B * NQ;
    float* c_last = h_last + HID;

    const int nblocks = B / TM;
    drqn_main<<<nblocks, 128, 0, stream>>>(obs, pact, cell, b1, b2,
                                           W1b, Wihb, W2b, hvec,
                                           q_out, h_last, c_last, nblocks);
}
